// DCHF_Block_55190329754132
// MI455X (gfx1250) — matrix operand order repaired and checked by reading, not compiled, not run on hardware
//
#include <hip/hip_runtime.h>
#include <math.h>

// ---------------------------------------------------------------------------
// MI455X / gfx1250 implementation of the DCHF block.
// Matrix-shaped work (all 1x1 convs + the big 112->64 3x3 "redu" conv) runs on
// v_wmma_f32_16x16x32_f16 (wave32, f32 accumulate), two independent WMMAs per
// K-chunk per wave (16x32 output tile). Operand gathers are branch-free:
// clamped-address unconditional loads + value selects, so the K-loop issues
// batched global loads instead of serialized exec-masked ones.
// ---------------------------------------------------------------------------

typedef __attribute__((ext_vector_type(16))) _Float16 v16h;
typedef __attribute__((ext_vector_type(8)))  float    v8f;

#define BATCH 2
#define CH    64
#define HH    96
#define WW    320
#define DD    48
#define HWPIX (HH * WW)          // 30720
#define CQ    16

__device__ __forceinline__ float dsigmoid(float x) { return 1.f / (1.f + __expf(-x)); }
__device__ __forceinline__ int   imin(int a, int b) { return a < b ? a : b; }

// ---------------------------------------------------------------------------
// Generic 1x1-conv GEMM:  Y[b,m,p] = act( sum_k W[m,k] * X[b,k,p] + bias[m] )
// One wave per 16(M) x 32(N) output tile: one A fragment, two B fragments,
// two WMMAs per K-chunk. K/M padded via clamped loads + selects (no branches).
// act: 0=none, 1=relu, 2=sigmoid
// ---------------------------------------------------------------------------
__global__ void k_gemm1x1(const float* __restrict__ X, const float* __restrict__ Wm,
                          const float* __restrict__ bias, float* __restrict__ Y,
                          int M, int K, int P, int nTiles2, int act)
{
    int wave   = (int)((blockIdx.x * (size_t)blockDim.x + threadIdx.x) >> 5);
    int tilesM = (M + 15) >> 4;
    if (wave >= tilesM * nTiles2) return;

    int tm = wave % tilesM;
    int tn = wave / tilesM;
    int lane = threadIdx.x & 31;
    int hi = lane >> 4;          // 0: lanes 0-15, 1: lanes 16-31
    int lo = lane & 15;

    int rowc = imin(tm * 16 + lo, M - 1);   // clamped A row (rows >= M only feed
                                            // D rows we never store)
    int n0 = tn * 32 + lo;                  // pixel of first N-tile
    int n1 = n0 + 16;                       // pixel of second N-tile
    int b0 = n0 / P, p0 = n0 - b0 * P;
    int b1 = n1 / P, p1 = n1 - b1 * P;
    const float* Xb0 = X + (size_t)b0 * K * P + p0;
    const float* Xb1 = X + (size_t)b1 * K * P + p1;
    const float* Wr  = Wm + (size_t)rowc * K;

    v8f acc0 = {0.f, 0.f, 0.f, 0.f, 0.f, 0.f, 0.f, 0.f};
    v8f acc1 = {0.f, 0.f, 0.f, 0.f, 0.f, 0.f, 0.f, 0.f};
    int nk = (K + 31) >> 5;
    for (int kc = 0; kc < nk; ++kc) {
        int k0 = kc << 5;
        if (kc + 1 < nk) {
            __builtin_prefetch(&Xb0[(size_t)(k0 + 32) * P], 0, 3);
            __builtin_prefetch(&Xb1[(size_t)(k0 + 32) * P], 0, 3);
        }
        v16h a, bm0, bm1;
#pragma unroll
        for (int j = 0; j < 16; ++j) {
            // A k order used here (same as B): lanes0-15 elem j -> K=j, lanes16-31 -> K=16+j
            int ka = k0 + (hi ? 16 : 0) + j;
            float av = Wr[imin(ka, K - 1)];          // unconditional load
            av = (ka < K) ? av : 0.f;                // v_cndmask
            a[j] = (_Float16)av;
            // B k order used here (same as A): lanes0-15 elem j -> K=j, lanes16-31 -> K=16+j
            int kb  = k0 + (hi ? 16 : 0) + j;
            int kbc = imin(kb, K - 1);
            float bv0 = Xb0[(size_t)kbc * P];
            float bv1 = Xb1[(size_t)kbc * P];
            bv0 = (kb < K) ? bv0 : 0.f;
            bv1 = (kb < K) ? bv1 : 0.f;
            bm0[j] = (_Float16)bv0;
            bm1[j] = (_Float16)bv1;
        }
        acc0 = __builtin_amdgcn_wmma_f32_16x16x32_f16(false, a, false, bm0,
                                                      (short)0, acc0, false, false);
        acc1 = __builtin_amdgcn_wmma_f32_16x16x32_f16(false, a, false, bm1,
                                                      (short)0, acc1, false, false);
    }
#pragma unroll
    for (int r = 0; r < 8; ++r) {
        int m = tm * 16 + r + hi * 8;   // f32 D layout: VGPR r holds M=r (+8 for hi lanes)
        if (m < M) {
            float bb = bias[m];
            float v0 = acc0[r] + bb;
            float v1 = acc1[r] + bb;
            if (act == 1)      { v0 = fmaxf(v0, 0.f);  v1 = fmaxf(v1, 0.f); }
            else if (act == 2) { v0 = dsigmoid(v0);    v1 = dsigmoid(v1); }
            Y[((size_t)b0 * M + m) * P + p0] = v0;
            Y[((size_t)b1 * M + m) * P + p1] = v1;
        }
    }
}

// ---------------------------------------------------------------------------
// redu conv: implicit GEMM, M=64, K=112*9=1008 (reflect pad 1), N = B*H*W.
// B operand gathered from fused[64ch] ++ norm_cost[48ch] with reflect indexing.
// One wave per 16x32 output tile, two WMMAs per K-chunk. Channel-split source
// chosen by pointer select (no branch).
// ---------------------------------------------------------------------------
__global__ void k_redu_wmma(const float* __restrict__ fus, const float* __restrict__ nc,
                            const float* __restrict__ Wm, const float* __restrict__ bias,
                            float* __restrict__ Y)
{
    const int KT = 112 * 9;  // 1008
    int wave = (int)((blockIdx.x * (size_t)blockDim.x + threadIdx.x) >> 5);
    const int tilesM  = 4;                       // M=64
    const int nTiles2 = (BATCH * HWPIX) / 32;    // 1920
    if (wave >= tilesM * nTiles2) return;

    int tm = wave % tilesM;
    int tn = wave / tilesM;
    int lane = threadIdx.x & 31;
    int hi = lane >> 4, lo = lane & 15;

    int row = tm * 16 + lo;
    int n0 = tn * 32 + lo;
    int n1 = n0 + 16;
    int b0 = n0 / HWPIX, p0 = n0 - b0 * HWPIX;
    int b1 = n1 / HWPIX, p1 = n1 - b1 * HWPIX;
    int h0 = p0 / WW, w0 = p0 - h0 * WW;
    int h1 = p1 / WW, w1 = p1 - h1 * WW;
    const float* Wr = Wm + (size_t)row * KT;

    v8f acc0 = {0.f, 0.f, 0.f, 0.f, 0.f, 0.f, 0.f, 0.f};
    v8f acc1 = {0.f, 0.f, 0.f, 0.f, 0.f, 0.f, 0.f, 0.f};
    for (int kc = 0; kc < 32; ++kc) {            // 32*32 = 1024 >= 1008
        int k0 = kc << 5;
        v16h a, bm0, bm1;
#pragma unroll
        for (int j = 0; j < 16; ++j) {
            int ka = k0 + (hi ? 16 : 0) + j;
            float av = Wr[imin(ka, KT - 1)];
            av = (ka < KT) ? av : 0.f;
            a[j] = (_Float16)av;

            int kb  = k0 + (hi ? 16 : 0) + j;
            int kbc = imin(kb, KT - 1);
            int c  = kbc / 9;
            int rr = kbc - 9 * c;
            int dh = rr / 3 - 1;
            int dw = rr - (rr / 3) * 3 - 1;
            // source plane: fused channel c, or norm_cost channel c-64 (ptr select)
            const float* pl0 = (c < CH) ? (fus + ((size_t)b0 * CH + c) * HWPIX)
                                        : (nc  + ((size_t)b0 * DD + (c - CH)) * HWPIX);
            const float* pl1 = (c < CH) ? (fus + ((size_t)b1 * CH + c) * HWPIX)
                                        : (nc  + ((size_t)b1 * DD + (c - CH)) * HWPIX);
            int hh0 = h0 + dh; hh0 = (hh0 < 0) ? 1 : ((hh0 >= HH) ? HH - 2 : hh0);
            int ww0 = w0 + dw; ww0 = (ww0 < 0) ? 1 : ((ww0 >= WW) ? WW - 2 : ww0);
            int hh1 = h1 + dh; hh1 = (hh1 < 0) ? 1 : ((hh1 >= HH) ? HH - 2 : hh1);
            int ww1 = w1 + dw; ww1 = (ww1 < 0) ? 1 : ((ww1 >= WW) ? WW - 2 : ww1);
            float bv0 = pl0[(size_t)hh0 * WW + ww0];
            float bv1 = pl1[(size_t)hh1 * WW + ww1];
            bv0 = (kb < KT) ? bv0 : 0.f;
            bv1 = (kb < KT) ? bv1 : 0.f;
            bm0[j] = (_Float16)bv0;
            bm1[j] = (_Float16)bv1;
        }
        acc0 = __builtin_amdgcn_wmma_f32_16x16x32_f16(false, a, false, bm0,
                                                      (short)0, acc0, false, false);
        acc1 = __builtin_amdgcn_wmma_f32_16x16x32_f16(false, a, false, bm1,
                                                      (short)0, acc1, false, false);
    }
#pragma unroll
    for (int r = 0; r < 8; ++r) {
        int m = tm * 16 + r + hi * 8;
        float bb = bias[m];
        Y[((size_t)b0 * CH + m) * HWPIX + p0] = acc0[r] + bb;
        Y[((size_t)b1 * CH + m) * HWPIX + p1] = acc1[r] + bb;
    }
}

// --------------------------- support kernels -------------------------------

__global__ void k_norminv(const float* __restrict__ q, const float* __restrict__ k,
                          float* __restrict__ qn, float* __restrict__ kn)
{
    size_t idx = blockIdx.x * (size_t)blockDim.x + threadIdx.x;
    if (idx >= (size_t)BATCH * HWPIX) return;
    int b = (int)(idx / HWPIX);
    int p = (int)(idx - (size_t)b * HWPIX);
    float sq = 0.f, sk = 0.f;
    for (int c = 0; c < CQ; ++c) {
        float a = q[((size_t)b * CQ + c) * HWPIX + p]; sq += a * a;
        float v = k[((size_t)b * CQ + c) * HWPIX + p]; sk += v * v;
    }
    qn[idx] = 1.f / (sqrtf(sq) + 1e-5f);
    kn[idx] = 1.f / (sqrtf(sk) + 1e-5f);
}

__global__ void k_costvol(const float* __restrict__ q, const float* __restrict__ k,
                          const float* __restrict__ qn, const float* __restrict__ kn,
                          float* __restrict__ cv)
{
    size_t idx = blockIdx.x * (size_t)blockDim.x + threadIdx.x;
    size_t total = (size_t)BATCH * DD * HWPIX;
    if (idx >= total) return;
    int w = (int)(idx % WW); size_t t = idx / WW;
    int h = (int)(t % HH); t /= HH;
    int d = (int)(t % DD); int b = (int)(t / DD);

    size_t hw = (size_t)h * WW + w;
    size_t o0 = ((size_t)(b * 2 + 0) * DD + d) * HWPIX + hw;
    size_t o1 = ((size_t)(b * 2 + 1) * DD + d) * HWPIX + hw;
    if (w < d) { cv[o0] = 0.f; cv[o1] = 0.f; return; }
    size_t hw2 = hw - d;
    float dot = 0.f, ad = 0.f;
    for (int c = 0; c < CQ; ++c) {
        float a = q[((size_t)b * CQ + c) * HWPIX + hw];
        float v = k[((size_t)b * CQ + c) * HWPIX + hw2];
        dot += a * v;
        ad  += fabsf(a - v);
    }
    cv[o0] = dot * qn[(size_t)b * HWPIX + hw] * kn[(size_t)b * HWPIX + hw2] * (1.f / 16.f);
    cv[o1] = ad * (1.f / 16.f);
}

__global__ void k_avgpool(const float* __restrict__ in, float* __restrict__ out,
                          int Hi, int Wi, int s)
{
    int Ho = Hi / s, Wo = Wi / s;
    size_t idx = blockIdx.x * (size_t)blockDim.x + threadIdx.x;
    size_t total = (size_t)BATCH * CH * Ho * Wo;
    if (idx >= total) return;
    int xo = (int)(idx % Wo); size_t t = idx / Wo;
    int yo = (int)(t % Ho);
    int bc = (int)(t / Ho);
    float sum = 0.f;
    for (int i = 0; i < s; ++i)
        for (int j = 0; j < s; ++j)
            sum += in[((size_t)bc * Hi + yo * s + i) * Wi + xo * s + j];
    out[idx] = sum / (float)(s * s);
}

// dual-input 3D conv, kernel 3, pad 1 (in2 may be null => plain conv)
__global__ void k_conv3d(const float* __restrict__ in1, const float* __restrict__ in2,
                         const float* __restrict__ wgt, float* __restrict__ out,
                         int C1, int C2, int Co, int Di, int Hi, int Wi, int st,
                         int Do, int Ho, int Wo, int leaky)
{
    size_t idx = blockIdx.x * (size_t)blockDim.x + threadIdx.x;
    size_t total = (size_t)BATCH * Co * Do * Ho * Wo;
    if (idx >= total) return;
    int x = (int)(idx % Wo); size_t t = idx / Wo;
    int y = (int)(t % Ho); t /= Ho;
    int z = (int)(t % Do); t /= Do;
    int co = (int)(t % Co);
    int b  = (int)(t / Co);
    int Cin = C1 + C2;
    float acc = 0.f;
    for (int ci = 0; ci < Cin; ++ci) {
        const float* src = (ci < C1)
            ? in1 + (size_t)(b * C1 + ci) * Di * Hi * Wi
            : in2 + (size_t)(b * C2 + (ci - C1)) * Di * Hi * Wi;
        const float* wp = wgt + ((size_t)co * Cin + ci) * 27;
        for (int kd = 0; kd < 3; ++kd) {
            int id = z * st + kd - 1; if (id < 0 || id >= Di) continue;
            for (int kh = 0; kh < 3; ++kh) {
                int ih = y * st + kh - 1; if (ih < 0 || ih >= Hi) continue;
                for (int kw = 0; kw < 3; ++kw) {
                    int iw = x * st + kw - 1; if (iw < 0 || iw >= Wi) continue;
                    acc += wp[kd * 9 + kh * 3 + kw] *
                           src[((size_t)id * Hi + ih) * Wi + iw];
                }
            }
        }
    }
    if (leaky) acc = acc > 0.f ? acc : 0.01f * acc;
    out[idx] = acc;
}

// transpose conv, kernel 3, stride 2, exact 2x upsample
__global__ void k_deconv3d(const float* __restrict__ in, const float* __restrict__ wgt,
                           float* __restrict__ out, int Ci, int Co,
                           int Di, int Hi, int Wi, int leaky)
{
    int Do = 2 * Di, Ho = 2 * Hi, Wo = 2 * Wi;
    size_t idx = blockIdx.x * (size_t)blockDim.x + threadIdx.x;
    size_t total = (size_t)BATCH * Co * Do * Ho * Wo;
    if (idx >= total) return;
    int x = (int)(idx % Wo); size_t t = idx / Wo;
    int y = (int)(t % Ho); t /= Ho;
    int z = (int)(t % Do); t /= Do;
    int co = (int)(t % Co);
    int b  = (int)(t / Co);
    float acc = 0.f;
    for (int ci = 0; ci < Ci; ++ci) {
        const float* src = in + (size_t)(b * Ci + ci) * Di * Hi * Wi;
        const float* wp = wgt + ((size_t)co * Ci + ci) * 27;
        for (int kd = 0; kd < 3; ++kd) {
            int tz = z + 1 - kd; if (tz < 0 || (tz & 1)) continue;
            int id = tz >> 1;    if (id >= Di) continue;
            for (int kh = 0; kh < 3; ++kh) {
                int ty = y + 1 - kh; if (ty < 0 || (ty & 1)) continue;
                int ih = ty >> 1;    if (ih >= Hi) continue;
                for (int kw = 0; kw < 3; ++kw) {
                    int tx = x + 1 - kw; if (tx < 0 || (tx & 1)) continue;
                    int iw = tx >> 1;    if (iw >= Wi) continue;
                    acc += wp[kd * 9 + kh * 3 + kw] *
                           src[((size_t)id * Hi + ih) * Wi + iw];
                }
            }
        }
    }
    if (leaky) acc = acc > 0.f ? acc : 0.01f * acc;
    out[idx] = acc;
}

// vol[b,c,d,p] *= gate[b,c,p]
__global__ void k_volmul(float* __restrict__ vol, const float* __restrict__ gate,
                         int C, int Dd, int P)
{
    size_t idx = blockIdx.x * (size_t)blockDim.x + threadIdx.x;
    size_t total = (size_t)BATCH * C * Dd * P;
    if (idx >= total) return;
    int p = (int)(idx % P); size_t t = idx / P;
    t /= Dd;                       // skip d
    int c = (int)(t % C);
    int b = (int)(t / C);
    vol[idx] *= gate[((size_t)b * C + c) * P + p];
}

__global__ void k_softmax48(const float* __restrict__ cvo, float* __restrict__ nc)
{
    size_t idx = blockIdx.x * (size_t)blockDim.x + threadIdx.x;
    if (idx >= (size_t)BATCH * HWPIX) return;
    int b = (int)(idx / HWPIX);
    int p = (int)(idx - (size_t)b * HWPIX);
    float v[DD];
    float mx = -1e30f;
    for (int d = 0; d < DD; ++d) {
        v[d] = cvo[((size_t)b * DD + d) * HWPIX + p];
        mx = fmaxf(mx, v[d]);
    }
    float s = 0.f;
    for (int d = 0; d < DD; ++d) { v[d] = __expf(v[d] - mx); s += v[d]; }
    float inv = 1.f / s;
    for (int d = 0; d < DD; ++d)
        nc[((size_t)b * DD + d) * HWPIX + p] = v[d] * inv;
}

__global__ void k_fusion(const float* __restrict__ tf, const float* __restrict__ cf,
                         const float* __restrict__ faw, const float* __restrict__ fab,
                         float* __restrict__ fus)
{
    size_t idx = blockIdx.x * (size_t)blockDim.x + threadIdx.x;
    if (idx >= (size_t)BATCH * HWPIX) return;
    int b = (int)(idx / HWPIX);
    int p = (int)(idx - (size_t)b * HWPIX);
    float s = fab[0];
    for (int c = 0; c < CH; ++c) s += faw[c]      * tf[((size_t)b * CH + c) * HWPIX + p];
    for (int c = 0; c < CH; ++c) s += faw[CH + c] * cf[((size_t)b * CH + c) * HWPIX + p];
    float al = dsigmoid(s);
    for (int c = 0; c < CH; ++c) {
        size_t o = ((size_t)b * CH + c) * HWPIX + p;
        fus[o] = al * tf[o] + (1.f - al) * cf[o];
    }
}

__global__ void k_sereduce(const float* __restrict__ xr, float* __restrict__ se_raw)
{
    __shared__ float sm[256];
    int bc = blockIdx.x;                  // 0 .. B*C-1
    const float* src = xr + (size_t)bc * HWPIX;
    float s = 0.f;
    for (int i = threadIdx.x; i < HWPIX; i += 256) s += src[i];
    sm[threadIdx.x] = s;
    __syncthreads();
    for (int o = 128; o > 0; o >>= 1) {
        if ((int)threadIdx.x < o) sm[threadIdx.x] += sm[threadIdx.x + o];
        __syncthreads();
    }
    if (threadIdx.x == 0) se_raw[bc] = sm[0] / (float)HWPIX;
}

__global__ void k_semlp(const float* __restrict__ se_raw, const float* __restrict__ w1,
                        const float* __restrict__ w2, float* __restrict__ se)
{
    int t = threadIdx.x;
    if (t >= BATCH * CH) return;
    int b = t >> 6, c = t & 63;
    float acc = 0.f;
    for (int j = 0; j < 4; ++j) {
        float h = 0.f;
        for (int ci = 0; ci < CH; ++ci) h += se_raw[b * CH + ci] * w1[j * CH + ci];
        h = fmaxf(h, 0.f);
        acc += h * w2[c * 4 + j];
    }
    se[t] = dsigmoid(acc);
}

__global__ void k_final_elu(const float* __restrict__ xr, const float* __restrict__ se,
                            float* __restrict__ out)
{
    size_t idx = blockIdx.x * (size_t)blockDim.x + threadIdx.x;
    size_t total = (size_t)BATCH * CH * HWPIX;
    if (idx >= total) return;
    int c = (int)((idx / HWPIX) % CH);
    int b = (int)(idx / ((size_t)CH * HWPIX));
    float v = xr[idx] * se[b * CH + c];
    out[idx] = v > 0.f ? v : (__expf(v) - 1.f);
}

// ---------------------------------------------------------------------------

#define CEILDIV(a, b) (((a) + (b) - 1) / (b))

extern "C" void kernel_launch(void* const* d_in, const int* in_sizes, int n_in,
                              void* d_out, int out_size, void* d_ws, size_t ws_size,
                              hipStream_t stream)
{
    (void)in_sizes; (void)n_in; (void)out_size; (void)ws_size;

    const float* t_feat = (const float*)d_in[0];
    const float* s_feat = (const float*)d_in[1];
    const float* wq = (const float*)d_in[2];   const float* bq = (const float*)d_in[3];
    const float* wk = (const float*)d_in[4];   const float* bk = (const float*)d_in[5];
    const float* c1a = (const float*)d_in[6];  const float* c1b = (const float*)d_in[7];
    const float* c2a = (const float*)d_in[8];  const float* c2b = (const float*)d_in[9];
    const float* c3a = (const float*)d_in[10]; const float* c3b = (const float*)d_in[11];
    const float* u3 = (const float*)d_in[12];  const float* agg1 = (const float*)d_in[13];
    const float* u2 = (const float*)d_in[14];  const float* agg0 = (const float*)d_in[15];
    const float* u1 = (const float*)d_in[16];
    const float* fa8_w1 = (const float*)d_in[17];  const float* fa8_b1 = (const float*)d_in[18];
    const float* fa8_w2 = (const float*)d_in[19];  const float* fa8_b2 = (const float*)d_in[20];
    const float* fa16_w1 = (const float*)d_in[21]; const float* fa16_b1 = (const float*)d_in[22];
    const float* fa16_w2 = (const float*)d_in[23]; const float* fa16_b2 = (const float*)d_in[24];
    const float* fa32_w1 = (const float*)d_in[25]; const float* fa32_b1 = (const float*)d_in[26];
    const float* fa32_w2 = (const float*)d_in[27]; const float* fa32_b2 = (const float*)d_in[28];
    const float* fu16_w1 = (const float*)d_in[29]; const float* fu16_b1 = (const float*)d_in[30];
    const float* fu16_w2 = (const float*)d_in[31]; const float* fu16_b2 = (const float*)d_in[32];
    const float* fu8_w1 = (const float*)d_in[33];  const float* fu8_b1 = (const float*)d_in[34];
    const float* fu8_w2 = (const float*)d_in[35];  const float* fu8_b2 = (const float*)d_in[36];
    const float* ca_w1 = (const float*)d_in[37];   const float* ca_b1 = (const float*)d_in[38];
    const float* ca_w2 = (const float*)d_in[39];   const float* ca_b2 = (const float*)d_in[40];
    const float* ce_w = (const float*)d_in[41];    const float* ce_b = (const float*)d_in[42];
    const float* fa_w = (const float*)d_in[43];    const float* fa_b = (const float*)d_in[44];
    const float* rd_w = (const float*)d_in[45];    const float* rd_b = (const float*)d_in[46];
    const float* se_w1 = (const float*)d_in[47];   const float* se_w2 = (const float*)d_in[48];

    float* outp = (float*)d_out;
    float* nc   = outp + (size_t)BATCH * CH * HWPIX;   // norm_cost region of d_out

    // workspace arena (floats)
    float* ws = (float*)d_ws;
    size_t off = 0;
    float* q      = ws + off; off += (size_t)BATCH * CQ * HWPIX;   // 983040
    float* k      = ws + off; off += (size_t)BATCH * CQ * HWPIX;
    float* qninv  = ws + off; off += (size_t)BATCH * HWPIX;
    float* kninv  = ws + off; off += (size_t)BATCH * HWPIX;
    float* cv     = ws + off; off += (size_t)BATCH * 2 * DD * HWPIX;
    float* im1    = ws + off; off += (size_t)BATCH * CH * 7680;
    float* im2    = ws + off; off += (size_t)BATCH * CH * 1920;
    float* im3    = ws + off; off += (size_t)BATCH * CH * 480;
    float* t1     = ws + off; off += (size_t)BATCH * 4 * 24 * 7680;
    float* c1     = ws + off; off += (size_t)BATCH * 4 * 24 * 7680;
    float* t2     = ws + off; off += (size_t)BATCH * 8 * 12 * 1920;
    float* c2     = ws + off; off += (size_t)BATCH * 8 * 12 * 1920;
    float* t3     = ws + off; off += (size_t)BATCH * 16 * 6 * 480;
    float* c3     = ws + off; off += (size_t)BATCH * 16 * 6 * 480;
    float* d2     = ws + off; off += (size_t)BATCH * 8 * 12 * 1920;
    float* x2     = ws + off; off += (size_t)BATCH * 8 * 12 * 1920;
    float* d1     = ws + off; off += (size_t)BATCH * 4 * 24 * 7680;
    float* x1     = ws + off; off += (size_t)BATCH * 4 * 24 * 7680;
    float* cvo    = ws + off; off += (size_t)BATCH * DD * HWPIX;
    float* atmp   = ws + off; off += (size_t)BATCH * 32 * HWPIX / 4 + (size_t)BATCH * 8 * HWPIX;
    float* gate   = ws + off; off += (size_t)BATCH * HWPIX;
    float* cfeat  = ws + off; off += (size_t)BATCH * CH * HWPIX;
    float* fusedb = ws + off; off += (size_t)BATCH * CH * HWPIX;
    float* xr     = ws + off; off += (size_t)BATCH * CH * HWPIX;
    float* se_raw = ws + off; off += 128;
    float* se     = ws + off; off += 128;

    auto gemm = [&](const float* X, const float* Wp, const float* bp, float* Y,
                    int M, int K, int P, int act) {
        int tilesM  = (M + 15) / 16;
        int nTiles2 = BATCH * P / 32;
        size_t waves = (size_t)tilesM * nTiles2;
        int blocks = (int)CEILDIV(waves * 32, 256);
        k_gemm1x1<<<blocks, 256, 0, stream>>>(X, Wp, bp, Y, M, K, P, nTiles2, act);
    };

    // ---- q / k projections + norms ----
    gemm(t_feat, wq, bq, q, 16, 64, HWPIX, 0);
    gemm(s_feat, wk, bk, k, 16, 64, HWPIX, 0);
    k_norminv<<<CEILDIV(BATCH * HWPIX, 256), 256, 0, stream>>>(q, k, qninv, kninv);

    // ---- pooled images ----
    k_avgpool<<<CEILDIV(BATCH * CH * 7680, 256), 256, 0, stream>>>(t_feat, im1, HH, WW, 2);
    k_avgpool<<<CEILDIV(BATCH * CH * 1920, 256), 256, 0, stream>>>(t_feat, im2, HH, WW, 4);
    k_avgpool<<<CEILDIV(BATCH * CH * 480, 256), 256, 0, stream>>>(t_feat, im3, HH, WW, 8);

    // ---- cost volume [B,2,D,H,W] ----
    k_costvol<<<CEILDIV((size_t)BATCH * DD * HWPIX, 256), 256, 0, stream>>>(q, k, qninv, kninv, cv);

    // ---- encoder ----
    k_conv3d<<<CEILDIV((size_t)BATCH * 4 * 24 * 7680, 256), 256, 0, stream>>>(
        cv, nullptr, c1a, t1, 2, 0, 4, 48, 96, 320, 2, 24, 48, 160, 1);
    k_conv3d<<<CEILDIV((size_t)BATCH * 4 * 24 * 7680, 256), 256, 0, stream>>>(
        t1, nullptr, c1b, c1, 4, 0, 4, 24, 48, 160, 1, 24, 48, 160, 1);
    gemm(im1, fa8_w1, fa8_b1, atmp, 32, 64, 7680, 1);
    gemm(atmp, fa8_w2, fa8_b2, gate, 4, 32, 7680, 2);
    k_volmul<<<CEILDIV((size_t)BATCH * 4 * 24 * 7680, 256), 256, 0, stream>>>(c1, gate, 4, 24, 7680);

    k_conv3d<<<CEILDIV((size_t)BATCH * 8 * 12 * 1920, 256), 256, 0, stream>>>(
        c1, nullptr, c2a, t2, 4, 0, 8, 24, 48, 160, 2, 12, 24, 80, 1);
    k_conv3d<<<CEILDIV((size_t)BATCH * 8 * 12 * 1920, 256), 256, 0, stream>>>(
        t2, nullptr, c2b, c2, 8, 0, 8, 12, 24, 80, 1, 12, 24, 80, 1);
    gemm(im2, fa16_w1, fa16_b1, atmp, 32, 64, 1920, 1);
    gemm(atmp, fa16_w2, fa16_b2, gate, 8, 32, 1920, 2);
    k_volmul<<<CEILDIV((size_t)BATCH * 8 * 12 * 1920, 256), 256, 0, stream>>>(c2, gate, 8, 12, 1920);

    k_conv3d<<<CEILDIV((size_t)BATCH * 16 * 6 * 480, 256), 256, 0, stream>>>(
        c2, nullptr, c3a, t3, 8, 0, 16, 12, 24, 80, 2, 6, 12, 40, 1);
    k_conv3d<<<CEILDIV((size_t)BATCH * 16 * 6 * 480, 256), 256, 0, stream>>>(
        t3, nullptr, c3b, c3, 16, 0, 16, 6, 12, 40, 1, 6, 12, 40, 1);
    gemm(im3, fa32_w1, fa32_b1, atmp, 32, 64, 480, 1);
    gemm(atmp, fa32_w2, fa32_b2, gate, 16, 32, 480, 2);
    k_volmul<<<CEILDIV((size_t)BATCH * 16 * 6 * 480, 256), 256, 0, stream>>>(c3, gate, 16, 6, 480);

    // ---- decoder ----
    k_deconv3d<<<CEILDIV((size_t)BATCH * 8 * 12 * 1920, 256), 256, 0, stream>>>(
        c3, u3, d2, 16, 8, 6, 12, 40, 1);
    k_conv3d<<<CEILDIV((size_t)BATCH * 8 * 12 * 1920, 256), 256, 0, stream>>>(
        d2, c2, agg1, x2, 8, 8, 8, 12, 24, 80, 1, 12, 24, 80, 1);
    gemm(im2, fu16_w1, fu16_b1, atmp, 32, 64, 1920, 1);
    gemm(atmp, fu16_w2, fu16_b2, gate, 8, 32, 1920, 2);
    k_volmul<<<CEILDIV((size_t)BATCH * 8 * 12 * 1920, 256), 256, 0, stream>>>(x2, gate, 8, 12, 1920);

    k_deconv3d<<<CEILDIV((size_t)BATCH * 4 * 24 * 7680, 256), 256, 0, stream>>>(
        x2, u2, d1, 8, 4, 12, 24, 80, 1);
    k_conv3d<<<CEILDIV((size_t)BATCH * 4 * 24 * 7680, 256), 256, 0, stream>>>(
        d1, c1, agg0, x1, 4, 4, 4, 24, 48, 160, 1, 24, 48, 160, 1);
    gemm(im1, fu8_w1, fu8_b1, atmp, 32, 64, 7680, 1);
    gemm(atmp, fu8_w2, fu8_b2, gate, 4, 32, 7680, 2);
    k_volmul<<<CEILDIV((size_t)BATCH * 4 * 24 * 7680, 256), 256, 0, stream>>>(x1, gate, 4, 24, 7680);

    k_deconv3d<<<CEILDIV((size_t)BATCH * DD * HWPIX, 256), 256, 0, stream>>>(
        x1, u1, cvo, 4, 1, 24, 48, 160, 0);                     // no activation

    // ---- channelAtt gated by q, softmax over disparity ----
    gemm(q, ca_w1, ca_b1, atmp, 8, 16, HWPIX, 1);
    gemm(atmp, ca_w2, ca_b2, gate, 1, 8, HWPIX, 2);
    k_volmul<<<CEILDIV((size_t)BATCH * DD * HWPIX, 256), 256, 0, stream>>>(cvo, gate, 1, DD, HWPIX);
    k_softmax48<<<CEILDIV(BATCH * HWPIX, 256), 256, 0, stream>>>(cvo, nc);

    // ---- cost expand + fusion ----
    gemm(nc, ce_w, ce_b, cfeat, 64, 48, HWPIX, 0);
    k_fusion<<<CEILDIV(BATCH * HWPIX, 256), 256, 0, stream>>>(t_feat, cfeat, fa_w, fa_b, fusedb);

    // ---- redu conv (WMMA implicit GEMM) + SE + ELU ----
    {
        size_t waves = (size_t)4 * (BATCH * HWPIX / 32);
        int blocks = (int)CEILDIV(waves * 32, 256);
        k_redu_wmma<<<blocks, 256, 0, stream>>>(fusedb, nc, rd_w, rd_b, xr);
    }
    k_sereduce<<<BATCH * CH, 256, 0, stream>>>(xr, se_raw);
    k_semlp<<<1, 128, 0, stream>>>(se_raw, se_w1, se_w2, se);
    k_final_elu<<<CEILDIV((size_t)BATCH * CH * HWPIX, 256), 256, 0, stream>>>(xr, se, outp);
}
